// Attention_9431748182241
// MI455X (gfx1250) — matrix operand order repaired and checked by reading, not compiled, not run on hardware
//
#include <hip/hip_runtime.h>

// ---------------------------------------------------------------------------
// MI455X (gfx1250) implementation of multi-scale depthwise channel attention.
// wave32, WMMA 16x16x32 bf16 for GEMM work, Tensor Data Mover for tile staging.
// ---------------------------------------------------------------------------

typedef __attribute__((ext_vector_type(16))) __bf16 v16bf;
typedef __attribute__((ext_vector_type(8)))  float  v8f;
typedef __attribute__((ext_vector_type(4)))  int    v4i;
typedef __attribute__((ext_vector_type(8)))  int    v8i;

#define WMMA_BF16(a, b, c) \
  __builtin_amdgcn_wmma_f32_16x16x32_bf16(false, (a), false, (b), (short)0, (c), false, false)

// ---------------------------------------------------------------------------
// Tensor Data Mover: async 2D tile load Global -> LDS (f32 elements).
// D# group0: count=1 | lds_addr | global_addr[56:0] | type=2.
// D# group1: data_size=4B, pad_enable (1 DWORD pad every 16 DWORDs so a
// [tile_d1][tile_d0] f32 tile lands as rows of (tile_d0+1) DWORDs -> bank
// destaggered), tensor/tile dims, dim0 stride. 2-group form (2D tensor).
// Issue from a single wave; completion via s_wait_tensorcnt + barrier.
// ---------------------------------------------------------------------------
__device__ __forceinline__ void tdm_load_2d_f32(unsigned lds_off, const float* gptr,
                                                unsigned tile_d0, unsigned tile_d1,
                                                unsigned long long tensor_d0,
                                                unsigned tensor_d1,
                                                unsigned long long stride_d0) {
  unsigned long long ga = (unsigned long long)(uintptr_t)gptr;
  v4i g0;
  g0[0] = 1;                                                  // count=1 (valid)
  g0[1] = (int)lds_off;                                       // LDS byte address
  g0[2] = (int)(ga & 0xffffffffull);                          // global_addr lo
  g0[3] = (int)(((ga >> 32) & 0x1ffffffull) | 0x80000000u);   // addr[56:32]|type=2
  v8i g1;
  g1[0] = (2 << 16) | (1 << 20) | (3 << 22);  // data_size=4B, pad_en, 16DW/pad 1DW
  g1[1] = (int)((tensor_d0 & 0xffffull) << 16);               // tensor_dim0[15:0]
  g1[2] = (int)(((tensor_d0 >> 16) & 0xffffull) |
                ((unsigned long long)(tensor_d1 & 0xffffu) << 16));
  g1[3] = (int)(((tensor_d1 >> 16) & 0xffffu) | (tile_d0 << 16));  // tile_dim0
  g1[4] = (int)(tile_d1 & 0xffffu);                           // tile_dim1
  g1[5] = (int)(stride_d0 & 0xffffffffull);                   // dim0 stride lo
  g1[6] = (int)((stride_d0 >> 32) & 0xffffull);               // dim0 stride hi
  g1[7] = 0;
  asm volatile("tensor_load_to_lds %0, %1" :: "s"(g0), "s"(g1) : "memory");
}

// Build A-fragment (16x32 bf16, M=lane%16, g=lane/16) from an f32 row pointer.
// Layout: lane holds K = 16g..16g+15 contiguously (matches B fragments).
__device__ __forceinline__ v16bf load_a_frag_f32(const float* __restrict__ rowp, int g) {
  v16bf a;
#pragma unroll
  for (int e = 0; e < 8; ++e) {
    a[e]     = (__bf16)rowp[16 * g + e];
    a[e + 8] = (__bf16)rowp[16 * g + 8 + e];
  }
  return a;
}

// B-fragment: lane holds 16 consecutive K values for its column.
__device__ __forceinline__ v16bf load_b_frag_lds(const __bf16* p) {
  v16bf b;
#pragma unroll
  for (int e = 0; e < 16; ++e) b[e] = p[e];
  return b;
}

// ---------------------------------------------------------------------------
// Kernel 1: xq[b, 384, N] = w_qkv[384,128] x x[b,128,N]   (N = 16384)
// Block = 16 pixels, 8 waves. x tile staged in LDS by the TDM as
// [128 ch][16 px + 1 pad] f32; B-fragments built with conflict-free ds_load_b32
// + cvt, A-fragments streamed from the f32 weight rows.
// ---------------------------------------------------------------------------
__global__ __launch_bounds__(256) void k_qkv(const float* __restrict__ x,
                                             const float* __restrict__ w_qkv,
                                             float* __restrict__ xq) {
  __shared__ float xs[128 * 17 + 17];  // TDM-padded tile
  const int ptile = blockIdx.x, b = blockIdx.y;
  const int pbase = ptile * 16;
  const int tid = threadIdx.x;

  if (tid < 32) {  // wave 0 issues the DMA
    tdm_load_2d_f32((unsigned)(uintptr_t)&xs[0],
                    x + (((size_t)b * 128) << 14) + pbase,
                    /*tile_d0=*/16, /*tile_d1=*/128,
                    /*tensor_d0=*/16384, /*tensor_d1=*/128,
                    /*stride_d0=*/16384);
    __builtin_amdgcn_s_wait_tensorcnt(0);
  }
  __syncthreads();

  const int wave = tid >> 5, lane = tid & 31;
  const int nl = lane & 15, g = lane >> 4;

  v16bf bf[4];
#pragma unroll
  for (int ks = 0; ks < 4; ++ks) {
    v16bf t;
#pragma unroll
    for (int e = 0; e < 16; ++e)
      t[e] = (__bf16)xs[(ks * 32 + 16 * g + e) * 17 + nl];
    bf[ks] = t;
  }

  for (int mt = wave; mt < 24; mt += 8) {
    v8f acc = {};
    const float* wrow = w_qkv + (mt * 16 + nl) * 128;
#pragma unroll
    for (int ks = 0; ks < 4; ++ks) {
      v16bf af = load_a_frag_f32(wrow + ks * 32, g);
      acc = WMMA_BF16(af, bf[ks], acc);
    }
#pragma unroll
    for (int r = 0; r < 8; ++r) {
      int o = mt * 16 + r + 8 * g;  // C/D layout: M = r + 8*(lane/16), N = lane%16
      xq[((b * 384 + o) << 14) + pbase + nl] = acc[r];
    }
  }
}

// ---------------------------------------------------------------------------
// Kernel 2: fused multi-scale depthwise conv + q/k/v projection.
// Tile: 8 rows x 16 cols (128 pixels). Per group (q/k/v): loop 4 chunks of 32
// channels, load halo into LDS (VALU path: zero-padded borders), compute
// dw3/dw5/dw7 into a [128px][32k] bf16 buffer, and immediately consume it as
// one WMMA K-step of the 128x384 projection GEMM.
// ---------------------------------------------------------------------------
__global__ __launch_bounds__(256) void k_dwproj(
    const float* __restrict__ xq,
    const float* __restrict__ w_dw3, const float* __restrict__ w_dw5,
    const float* __restrict__ w_dw7,
    const float* __restrict__ w_q, const float* __restrict__ w_k,
    const float* __restrict__ w_v,
    float* __restrict__ qb, float* __restrict__ kbuf, float* __restrict__ vb) {
  __shared__ float hin[14][22][33];          // halo: 14 rows x 22 cols x 32 ch (+pad)
  __shared__ float wl3[32][9];
  __shared__ float wl5[32][25];
  __shared__ float wl7[32][49];
  __shared__ __bf16 dsb[128][34];            // dwconv output chunk: [pixel][k-local]

  const int wbase = blockIdx.x * 16, hbase = blockIdx.y * 8, b = blockIdx.z;
  const int tid = threadIdx.x, wave = tid >> 5, lane = tid & 31;
  const int nl = lane & 15, g16 = lane >> 4;
  const v8f vzero = {};

  for (int grp = 0; grp < 3; ++grp) {
    const float* wp = (grp == 0) ? w_q : ((grp == 1) ? w_k : w_v);
    float* ob = (grp == 0) ? qb : ((grp == 1) ? kbuf : vb);

    v8f acc[8];
#pragma unroll
    for (int pt = 0; pt < 8; ++pt) acc[pt] = vzero;

    for (int cc = 0; cc < 4; ++cc) {
      __syncthreads();  // previous users of hin / dsb done
      const int choff = grp * 128 + cc * 32;

      // Halo load (coalesced along columns), zero-padded at image borders.
      for (int idx = tid; idx < 32 * 14 * 22; idx += 256) {
        int ch = idx / 308, rem = idx - ch * 308;
        int r = rem / 22, c = rem - r * 22;
        int gh = hbase + r - 3, gw = wbase + c - 3;
        float v = 0.f;
        if ((unsigned)gh < 128u && (unsigned)gw < 128u)
          v = xq[((b * 384 + choff + ch) << 14) + (gh << 7) + gw];
        hin[r][c][ch] = v;
      }
      for (int idx = tid; idx < 32 * 9; idx += 256)
        wl3[idx / 9][idx % 9] = w_dw3[(choff + idx / 9) * 9 + idx % 9];
      for (int idx = tid; idx < 32 * 25; idx += 256)
        wl5[idx / 25][idx % 25] = w_dw5[(choff + idx / 25) * 25 + idx % 25];
      for (int idx = tid; idx < 32 * 49; idx += 256)
        wl7[idx / 49][idx % 49] = w_dw7[(choff + idx / 49) * 49 + idx % 49];
      __syncthreads();

#pragma unroll
      for (int s = 0; s < 3; ++s) {
        // Depthwise taps: thread -> (ch = lane, pixels wave + 8*i).
        for (int i = 0; i < 16; ++i) {
          int px = wave + 8 * i;
          int pr = px >> 4, pc = px & 15;
          float a = 0.f;
          if (s == 0) {
#pragma unroll
            for (int ty = 0; ty < 3; ++ty)
#pragma unroll
              for (int tx = 0; tx < 3; ++tx)
                a += hin[pr + 2 + ty][pc + 2 + tx][lane] * wl3[lane][ty * 3 + tx];
          } else if (s == 1) {
#pragma unroll
            for (int ty = 0; ty < 5; ++ty)
#pragma unroll
              for (int tx = 0; tx < 5; ++tx)
                a += hin[pr + 1 + ty][pc + 1 + tx][lane] * wl5[lane][ty * 5 + tx];
          } else {
#pragma unroll
            for (int ty = 0; ty < 7; ++ty)
#pragma unroll
              for (int tx = 0; tx < 7; ++tx)
                a += hin[pr + ty][pc + tx][lane] * wl7[lane][ty * 7 + tx];
          }
          dsb[px][lane] = (__bf16)a;
        }
        __syncthreads();

        // One WMMA K-step (K=32) of the projection GEMM; wave owns o-tile = wave.
        const int kg = s * 128 + cc * 32;
        const float* arow = wp + (wave * 16 + nl) * 384 + kg;
        v16bf af = load_a_frag_f32(arow, g16);
#pragma unroll
        for (int pt = 0; pt < 8; ++pt) {
          v16bf bfr = load_b_frag_lds(&dsb[pt * 16 + nl][16 * g16]);
          acc[pt] = WMMA_BF16(af, bfr, acc[pt]);
        }
        __syncthreads();  // before dsb is overwritten
      }
    }

    // Store this group's 128 x 128px tile.
#pragma unroll
    for (int pt = 0; pt < 8; ++pt) {
      int p = ((hbase + pt) << 7) + wbase + nl;
#pragma unroll
      for (int r = 0; r < 8; ++r) {
        int o = wave * 16 + r + 8 * g16;
        ob[((b * 128 + o) << 14) + p] = acc[pt][r];
      }
    }
  }
}

// ---------------------------------------------------------------------------
// Kernel 3: per (b,head): Gram G = q x k^T over N=16384 (WMMA, K split over
// 8 waves), sum-of-squares norms in f32 VALU, fold norms + temperature into
// logits, softmax -> attn[b,h,16,16].
// ---------------------------------------------------------------------------
__global__ __launch_bounds__(256) void k_attn(const float* __restrict__ qb,
                                              const float* __restrict__ kb,
                                              const float* __restrict__ temp,
                                              float* __restrict__ attn_out) {
  __shared__ float gsh[8][16][16];
  __shared__ float sqsh[8][32];
  __shared__ float sksh[8][32];
  __shared__ float nq[16], nk[16];
  __shared__ float lg[16][16];

  const int bh = blockIdx.x;  // b*8 + h
  const int tid = threadIdx.x, wave = tid >> 5, lane = tid & 31;
  const int nl = lane & 15, g = lane >> 4;
  const float* qrow = qb + ((size_t)(bh * 16 + nl) << 14);
  const float* krow = kb + ((size_t)(bh * 16 + nl) << 14);

  v8f acc = {};
  float sq = 0.f, sk = 0.f;
  for (int p = wave * 32; p < 16384; p += 256) {
    __builtin_prefetch(qrow + p + 2048, 0, 0);  // global_prefetch_b8
    __builtin_prefetch(krow + p + 2048, 0, 0);
    v16bf af, bf;
#pragma unroll
    for (int e = 0; e < 8; ++e) {
      float v0 = qrow[p + 16 * g + e];
      float v1 = qrow[p + 16 * g + 8 + e];
      af[e] = (__bf16)v0;
      af[e + 8] = (__bf16)v1;
      sq += v0 * v0 + v1 * v1;
    }
#pragma unroll
    for (int e = 0; e < 16; ++e) {
      float v = krow[p + 16 * g + e];
      bf[e] = (__bf16)v;
      sk += v * v;
    }
    acc = WMMA_BF16(af, bf, acc);
  }
#pragma unroll
  for (int r = 0; r < 8; ++r) gsh[wave][r + 8 * g][nl] = acc[r];
  sqsh[wave][lane] = sq;
  sksh[wave][lane] = sk;
  __syncthreads();

  const int c = tid >> 4, d = tid & 15;
  float gsum = 0.f;
  for (int w = 0; w < 8; ++w) gsum += gsh[w][c][d];
  if (tid < 16) {
    float aq = 0.f, ak = 0.f;
    for (int w = 0; w < 8; ++w) {
      aq += sqsh[w][tid] + sqsh[w][tid + 16];
      ak += sksh[w][tid] + sksh[w][tid + 16];
    }
    nq[tid] = sqrtf(aq);
    nk[tid] = sqrtf(ak);
  }
  __syncthreads();

  const float EPSV = 1e-12f;
  const float tval = temp[bh & 7];
  float l = gsum * tval / (fmaxf(nq[c], EPSV) * fmaxf(nk[d], EPSV));
  lg[c][d] = l;
  __syncthreads();

  float m = -3.4e38f;
  for (int e = 0; e < 16; ++e) m = fmaxf(m, lg[c][e]);
  float ssum = 0.f;
  for (int e = 0; e < 16; ++e) ssum += __expf(lg[c][e] - m);
  attn_out[bh * 256 + c * 16 + d] = __expf(l - m) / ssum;
}

// ---------------------------------------------------------------------------
// Kernel 4: fold w_out with block-diagonal attention into per-batch matrix:
// Wb[b][o][h*16+d] = sum_c w_out[o][h*16+c] * attn[b,h,c,d]
// ---------------------------------------------------------------------------
__global__ __launch_bounds__(256) void k_wb(const float* __restrict__ w_out,
                                            const float* __restrict__ attn,
                                            float* __restrict__ Wb) {
  int t = blockIdx.x * 256 + threadIdx.x;  // 8*128*128
  int b = t >> 14, rem = t & 16383;
  int o = rem >> 7, col = rem & 127;
  int h = col >> 4, d = col & 15;
  float s = 0.f;
#pragma unroll
  for (int c = 0; c < 16; ++c)
    s += w_out[(o << 7) + (h << 4) + c] * attn[(((b * 8 + h) * 16 + c) << 4) + d];
  Wb[t] = s;
}

// ---------------------------------------------------------------------------
// Kernel 5: out[b,128,N] = Wb[b] x v[b]   (combines attn*v and the final 1x1)
// v tile staged via TDM, same structure as kernel 1 with M=128.
// ---------------------------------------------------------------------------
__global__ __launch_bounds__(256) void k_out(const float* __restrict__ vb,
                                             const float* __restrict__ Wb,
                                             float* __restrict__ out) {
  __shared__ float xs[128 * 17 + 17];
  const int ptile = blockIdx.x, b = blockIdx.y;
  const int pbase = ptile * 16;
  const int tid = threadIdx.x;

  if (tid < 32) {
    tdm_load_2d_f32((unsigned)(uintptr_t)&xs[0],
                    vb + (((size_t)b * 128) << 14) + pbase,
                    /*tile_d0=*/16, /*tile_d1=*/128,
                    /*tensor_d0=*/16384, /*tensor_d1=*/128,
                    /*stride_d0=*/16384);
    __builtin_amdgcn_s_wait_tensorcnt(0);
  }
  __syncthreads();

  const int wave = tid >> 5, lane = tid & 31;
  const int nl = lane & 15, g = lane >> 4;

  v16bf bf[4];
#pragma unroll
  for (int ks = 0; ks < 4; ++ks) {
    v16bf t;
#pragma unroll
    for (int e = 0; e < 16; ++e)
      t[e] = (__bf16)xs[(ks * 32 + 16 * g + e) * 17 + nl];
    bf[ks] = t;
  }

  v8f acc = {};
  const float* wrow = Wb + (b * 128 + wave * 16 + nl) * 128;
#pragma unroll
  for (int ks = 0; ks < 4; ++ks) {
    v16bf af = load_a_frag_f32(wrow + ks * 32, g);
    acc = WMMA_BF16(af, bf[ks], acc);
  }
#pragma unroll
  for (int r = 0; r < 8; ++r) {
    int o = wave * 16 + r + 8 * g;
    out[((b * 128 + o) << 14) + pbase + nl] = acc[r];
  }
}

// ---------------------------------------------------------------------------
extern "C" void kernel_launch(void* const* d_in, const int* in_sizes, int n_in,
                              void* d_out, int out_size, void* d_ws, size_t ws_size,
                              hipStream_t stream) {
  (void)in_sizes; (void)n_in; (void)out_size; (void)ws_size;
  const float* x      = (const float*)d_in[0];
  const float* w_qkv  = (const float*)d_in[1];
  const float* w_dw3  = (const float*)d_in[2];
  const float* w_dw5  = (const float*)d_in[3];
  const float* w_dw7  = (const float*)d_in[4];
  const float* w_q    = (const float*)d_in[5];
  const float* w_k    = (const float*)d_in[6];
  const float* w_v    = (const float*)d_in[7];
  const float* w_out  = (const float*)d_in[8];
  const float* temp   = (const float*)d_in[9];
  float* out = (float*)d_out;

  // Workspace layout (f32 elements).
  float* ws   = (float*)d_ws;
  float* xq   = ws;                      // 8*384*16384 = 50331648
  float* qb   = xq + 50331648;           // 8*128*16384 = 16777216
  float* kb   = qb + 16777216;
  float* vb   = kb + 16777216;
  float* attn = vb + 16777216;           // 8*8*16*16 = 16384
  float* Wb   = attn + 16384;            // 8*128*128 = 131072

  k_qkv  <<<dim3(1024, 8), 256, 0, stream>>>(x, w_qkv, xq);
  k_dwproj<<<dim3(8, 16, 8), 256, 0, stream>>>(xq, w_dw3, w_dw5, w_dw7,
                                               w_q, w_k, w_v, qb, kb, vb);
  k_attn <<<dim3(64), 256, 0, stream>>>(qb, kb, temp, attn);
  k_wb   <<<dim3(512), 256, 0, stream>>>(w_out, attn, Wb);
  k_out  <<<dim3(1024, 8), 256, 0, stream>>>(vb, Wb, out);
}
